// CharacterLoss_38457137168846
// MI455X (gfx1250) — matrix operand order repaired and checked by reading, not compiled, not run on hardware
//
#include <hip/hip_runtime.h>
#include <cstdint>

// ---------------------------------------------------------------------------
// Problem constants (from the reference): data [8192,1024] f32, indices [4096],
// token_ids [8192] i32.
// ---------------------------------------------------------------------------
constexpr int D          = 1024;
constexpr int M          = 4096;
constexpr int TILE       = 128;
constexpr int TILES      = M / TILE;                // 32
constexpr int NBLOCKS    = TILES * (TILES + 1) / 2; // 528 (upper triangle)
constexpr int KSTEP      = 64;                      // K elems staged per stage
constexpr int NSTEP      = D / KSTEP;               // 16
constexpr int LDS_STRIDE = 72;   // bf16 elems/row: 144B rows -> 16B aligned,
                                 // bank stride 36 -> conflict-free columns
constexpr int BLK_ELEMS  = TILE * LDS_STRIDE;       // 9216 elems / block
constexpr int DYN_LDS_BYTES = 2 * 4 * BLK_ELEMS * 2; // 147456 B (2-stage x4)

typedef __attribute__((ext_vector_type(16))) __bf16       v16bf;
typedef __attribute__((ext_vector_type(8)))  float        v8f;
typedef __attribute__((ext_vector_type(4)))  unsigned int v4u;

struct Frag {
  union { v16bf v; v4u q[2]; };
};

// ---------------------------------------------------------------------------
// Kernel 1: gather rows, split fp32 -> bf16 hi + bf16 lo, exact fp32 norms,
// gather token ids. One block per output row.
// ---------------------------------------------------------------------------
__global__ void __launch_bounds__(256)
prep_kernel(const float* __restrict__ data, const int* __restrict__ token_ids,
            const int* __restrict__ indices, __bf16* __restrict__ Ahi,
            __bf16* __restrict__ Alo, float* __restrict__ norms,
            int* __restrict__ toks) {
  const int m   = blockIdx.x;
  const int t   = threadIdx.x;
  const int src = indices[m];
  const float* row = data + (size_t)src * D;
  __bf16* hi = Ahi + (size_t)m * D;
  __bf16* lo = Alo + (size_t)m * D;
  float s = 0.f;
  for (int d = t; d < D; d += 256) {
    float  x = row[d];
    __bf16 h = (__bf16)x;
    __bf16 l = (__bf16)(x - (float)h);
    hi[d] = h;
    lo[d] = l;
    s += x * x;
  }
  __shared__ float red[256];
  red[t] = s;
  __syncthreads();
  for (int off = 128; off > 0; off >>= 1) {
    if (t < off) red[t] += red[t + off];
    __syncthreads();
  }
  if (t == 0) {
    norms[m] = sqrtf(red[0]);
    toks[m]  = token_ids[src];
  }
}

// ---------------------------------------------------------------------------
// Fragment loaders following the CDNA5 16-bit WMMA VGPR layouts (ISA 7.12.2).
// A (16x32, MxK): lane<16 -> M=lane,    VGPR0-3: K=0..7,  VGPR4-7: K=16..23
//                 lane>=16 -> M=lane-16, VGPR0-3: K=8..15, VGPR4-7: K=24..31
// B (32x16, KxN): lane<16 -> N=lane,    VGPR0-3: K=0..7,  VGPR4-7: K=16..23
//                 lane>=16 -> N=lane-16, VGPR0-3: K=8..15, VGPR4-7: K=24..31
// (B = A^T for the Gram matrix, so both read along K of a staged A row.)
// Each half-fragment is 8 contiguous bf16 (16 B, 16 B-aligned) -> ds_load_b128.
// ---------------------------------------------------------------------------
__device__ inline Frag load_afrag(const __bf16* base, int rowBase, int lane,
                                  int kOff) {
  Frag f;
  const int half = lane >> 4, r = lane & 15;
  const __bf16* row = base + (rowBase + r) * LDS_STRIDE + kOff;
  const int kA = half ? 8 : 0;
  const int kB = half ? 24 : 16;
  f.q[0] = *(const v4u*)(row + kA);
  f.q[1] = *(const v4u*)(row + kB);
  return f;
}

__device__ inline Frag load_bfrag(const __bf16* base, int rowBase, int lane,
                                  int kOff) {
  Frag f;
  const int half = lane >> 4, n = lane & 15;
  const __bf16* row = base + (rowBase + n) * LDS_STRIDE + kOff;
  const int k = half ? 8 : 0;
  f.q[0] = *(const v4u*)(row + k);
  f.q[1] = *(const v4u*)(row + k + 16);
  return f;
}

__device__ inline v8f wmma_bf16(const Frag& a, const Frag& b, v8f c) {
  return __builtin_amdgcn_wmma_f32_16x16x32_bf16(
      false, a.v, false, b.v, (short)0, c, false, false);
}

// Issue one stage of async global->LDS copies: 4 blocks of 128 rows x 64 bf16
// (128 B/row) = 1024 x 16 B chunks per block; 256 threads x 4 chunks; b128.
// Per wave: 16 async instructions per stage (pairs with s_wait_asynccnt 16).
__device__ inline void issue_stage(const __bf16* const* gsrc, int k0,
                                   __bf16* stage_base, int tid) {
#pragma unroll
  for (int b = 0; b < 4; ++b) {
#pragma unroll
    for (int h = 0; h < 4; ++h) {
      const int c   = tid + h * 256;       // 0..1023
      const int row = c >> 3, q = c & 7;   // q: 16B chunk within the row
      const __bf16* src = gsrc[b] + (size_t)row * D + k0 + q * 8;
      const uint32_t dst = (uint32_t)(uintptr_t)(
          stage_base + b * BLK_ELEMS + row * LDS_STRIDE + q * 8);
      asm volatile("global_load_async_to_lds_b128 %0, %1, off"
                   :: "v"(dst), "v"(src) : "memory");
    }
  }
}

// ---------------------------------------------------------------------------
// Kernel 2: fused triangular Gram GEMM + BCE loss epilogue.
// 256 threads = 8 waves; 128x128 output tile; wave w owns a 16-row strip.
// Double-buffered async LDS pipeline, 48 WMMAs per wave between barriers.
// Column sub-tiles processed in groups of 4 with term-major WMMA issue so
// same-accumulator WMMAs are spaced by 3 independent WMMAs (hazard window
// for bf16 WMMA is 1 NOP + 4 co-exec slots, ISA 7.12.1) -> no v_nop filler.
// hi/lo split: dot = hi*hi + hi*lo + lo*hi  (lo*lo dropped, ~2^-18 rel err).
// ---------------------------------------------------------------------------
__global__ void __launch_bounds__(256)
gemm_loss_kernel(const __bf16* __restrict__ Ahi, const __bf16* __restrict__ Alo,
                 const float* __restrict__ norms, const int* __restrict__ toks,
                 float* __restrict__ partial) {
  extern __shared__ __bf16 dynlds[];  // 2 stages x 4 blocks x 128x72 bf16
  __bf16* stage0 = dynlds;
  __bf16* stage1 = dynlds + 4 * BLK_ELEMS;

  // blockIdx -> (bi, bj) with bi <= bj (upper-triangular tile enumeration)
  int t  = blockIdx.x;
  int bi = 0;
  while (t >= (TILES - bi)) { t -= (TILES - bi); ++bi; }
  const int bj = bi + t;
  const int r0 = bi * TILE, c0 = bj * TILE;

  const int tid = threadIdx.x, wave = tid >> 5, lane = tid & 31;

  v8f acc[8];
#pragma unroll
  for (int j = 0; j < 8; ++j) acc[j] = (v8f){0, 0, 0, 0, 0, 0, 0, 0};

  const __bf16* gsrc[4] = {Ahi + (size_t)r0 * D, Alo + (size_t)r0 * D,
                           Ahi + (size_t)c0 * D, Alo + (size_t)c0 * D};

  issue_stage(gsrc, 0, stage0, tid);            // prologue: stage 0 -> buf 0

  for (int step = 0; step < NSTEP; ++step) {
    __bf16* cur = (step & 1) ? stage1 : stage0;
    __bf16* nxt = (step & 1) ? stage0 : stage1;
    if (step + 1 < NSTEP) {
      issue_stage(gsrc, (step + 1) * KSTEP, nxt, tid);
      // leave the 16 just-issued copies in flight; wait only for `cur`'s
      asm volatile("s_wait_asynccnt 16" ::: "memory");
    } else {
      asm volatile("s_wait_asynccnt 0" ::: "memory");
    }
    __syncthreads();

#pragma unroll
    for (int ks = 0; ks < 2; ++ks) {
      const int kOff = ks * 32;
      const Frag ahi = load_afrag(cur + 0 * BLK_ELEMS, wave * 16, lane, kOff);
      const Frag alo = load_afrag(cur + 1 * BLK_ELEMS, wave * 16, lane, kOff);
#pragma unroll
      for (int g = 0; g < 8; g += 4) {
        Frag bhi[4], blo[4];
#pragma unroll
        for (int u = 0; u < 4; ++u) {
          bhi[u] = load_bfrag(cur + 2 * BLK_ELEMS, (g + u) * 16, lane, kOff);
          blo[u] = load_bfrag(cur + 3 * BLK_ELEMS, (g + u) * 16, lane, kOff);
        }
        // term-major issue: 4 independent accumulator chains interleaved
#pragma unroll
        for (int u = 0; u < 4; ++u)
          acc[g + u] = wmma_bf16(ahi, bhi[u], acc[g + u]);
#pragma unroll
        for (int u = 0; u < 4; ++u)
          acc[g + u] = wmma_bf16(ahi, blo[u], acc[g + u]);
#pragma unroll
        for (int u = 0; u < 4; ++u)
          acc[g + u] = wmma_bf16(alo, bhi[u], acc[g + u]);
      }
    }
    __syncthreads();  // everyone done reading `cur` before it is re-filled
  }

  // Epilogue: C/D layout (ISA 7.12.2): lane<16 -> N=lane, M=v;
  // lane>=16 -> N=lane-16, M=v+8. Fuse cosine normalization + stable
  // BCE-with-logits + triangular weighting.
  const int half = lane >> 4, ln = lane & 15;
  float ni[8];
  int   ti[8];
#pragma unroll
  for (int v = 0; v < 8; ++v) {
    const int i = r0 + wave * 16 + half * 8 + v;
    ni[v] = norms[i];
    ti[v] = toks[i];
  }
  float lsum = 0.f;
#pragma unroll
  for (int j = 0; j < 8; ++j) {
    const int   col = c0 + j * 16 + ln;
    const float nj  = norms[col];
    const int   tj  = toks[col];
#pragma unroll
    for (int v = 0; v < 8; ++v) {
      const int   i   = r0 + wave * 16 + half * 8 + v;
      const float z   = acc[j][v] / fmaxf(ni[v] * nj, 1e-8f);
      const float tgt = (ti[v] == tj) ? 1.f : 0.f;
      const float la  = (z > 0.f) ? (z + log1pf(expf(-z))) : log1pf(expf(z));
      const float term = la - z * tgt;
      float w;
      if (bi != bj) w = 2.f;                       // mirrored tile
      else          w = (i < col) ? 2.f : (i == col ? 1.f : 0.f);
      lsum += w * term;
    }
  }

  __shared__ float red[256];
  red[tid] = lsum;
  __syncthreads();
  for (int off = 128; off > 0; off >>= 1) {
    if (tid < off) red[tid] += red[tid + off];
    __syncthreads();
  }
  if (tid == 0) partial[blockIdx.x] = red[0];
}

// ---------------------------------------------------------------------------
// Kernel 3: deterministic fixed-order final reduction, mean over M*M.
// ---------------------------------------------------------------------------
__global__ void finalize_kernel(const float* __restrict__ partial,
                                float* __restrict__ out, int n) {
  if (threadIdx.x == 0 && blockIdx.x == 0) {
    double s = 0.0;
    for (int i = 0; i < n; ++i) s += (double)partial[i];
    out[0] = (float)(s / ((double)M * (double)M));
  }
}

// ---------------------------------------------------------------------------
extern "C" void kernel_launch(void* const* d_in, const int* in_sizes, int n_in,
                              void* d_out, int out_size, void* d_ws,
                              size_t ws_size, hipStream_t stream) {
  const float* data      = (const float*)d_in[0];
  const int*   token_ids = (const int*)d_in[1];
  const int*   indices   = (const int*)d_in[2];
  float*       out       = (float*)d_out;

  char* ws = (char*)d_ws;
  const size_t half_sz = (size_t)M * D * sizeof(__bf16);  // 8 MB each
  __bf16* Ahi    = (__bf16*)(ws);
  __bf16* Alo    = (__bf16*)(ws + half_sz);
  float*  norms  = (float*)(ws + 2 * half_sz);
  int*    toks   = (int*)(ws + 2 * half_sz + M * sizeof(float));
  float*  part   = (float*)(ws + 2 * half_sz + M * sizeof(float) +
                            M * sizeof(int));

  prep_kernel<<<M, 256, 0, stream>>>(data, token_ids, indices, Ahi, Alo,
                                     norms, toks);
  gemm_loss_kernel<<<NBLOCKS, 256, DYN_LDS_BYTES, stream>>>(Ahi, Alo, norms,
                                                            toks, part);
  finalize_kernel<<<1, 64, 0, stream>>>(part, out, NBLOCKS);
}
